// FFMLayer_23338852286516
// MI455X (gfx1250) — matrix operand order repaired and checked by reading, not compiled, not run on hardware
//
#include <hip/hip_runtime.h>
#include <hip/hip_bf16.h>
#include <math.h>
#include <stdint.h>

// ---------------------------------------------------------------------------
// CDNA5 (gfx1250, wave32) implementation of the FFM layer.
// Heavy GEMMs (the two 768x768 MLPs per path) use v_wmma_f32_16x16x32_f16
// with async global->LDS staging (ASYNCcnt path). DFT stages (STFT 512-pt,
// rfft/irfft 196-pt, ISTFT) are table-driven f32 VALU with fused
// filter/gate/overlap-add (stop_gradient path, f32 precision required).
// ---------------------------------------------------------------------------

typedef _Float16 f16;
typedef __attribute__((ext_vector_type(16))) _Float16 v16h;
typedef __attribute__((ext_vector_type(8)))  float    v8f;

union V16H { v16h v; f16 e[16]; };
union V8F  { v8f  v; float e[8]; };

constexpr int B_   = 8;
constexpr int D_   = 768;
constexpr int S_   = 2048;
constexpr int NSEG = 9;
constexpr int NFREQ= 257;
constexpr int NPER = 512;
constexpr int STEPL= 256;
constexpr int PADL = 2560;          // 2048 + 2*256
constexpr int NIMG = 196;
constexpr int FIMG = 99;
constexpr int SE   = NFREQ * NSEG;  // 2313
constexpr float PI_F = 3.1415926f;  // matches reference constant

// ---------------------------------------------------------------------------
// f32 -> f16 weight conversion
__global__ void k_cvt(const float* __restrict__ x, f16* __restrict__ y, int n) {
  int i = blockIdx.x * blockDim.x + threadIdx.x;
  if (i < n) y[i] = (f16)x[i];
}

// ---------------------------------------------------------------------------
// 512-point DFT twiddle tables in both [n][f] (forward) and [f][n] (inverse)
// layouts; angle reduced mod 512 as an integer, evaluated in f64.
__global__ void k_tables(float* cNF, float* sNF, float* cFN, float* sFN) {
  int idx = blockIdx.x * blockDim.x + threadIdx.x;
  if (idx >= NPER * NFREQ) return;
  int n = idx / NFREQ, f = idx % NFREQ;
  int m = (n * f) & (NPER - 1);
  double ang = 6.283185307179586476925287 * (double)m / (double)NPER;
  float c = (float)cos(ang), s = (float)sin(ang);
  cNF[n * NFREQ + f] = c;  sNF[n * NFREQ + f] = s;
  cFN[f * NPER  + n] = c;  sFN[f * NPER  + n] = s;
}

// ---------------------------------------------------------------------------
// Combined ecg filter bank: cb[d][t*257+f] = coef0*bank[0,s,d] + coef1*bank[1,s,d]
// with s = f*9 + t (the reference's transpose/reshape index), complex.
// One block per d so the per-d stream the ISTFT kernel reads is contiguous.
__global__ void k_cbE(const float* __restrict__ bank, float* __restrict__ cb) {
  int d = blockIdx.x;
  float c0 = cosf(PI_F);          // cos((2*1-1)*PI/2*2)
  float c1 = cosf(3.0f * PI_F);   // cos((2*2-1)*PI/2*2)
  for (int i = threadIdx.x; i < SE; i += blockDim.x) {
    int t = i / NFREQ, f = i - t * NFREQ;
    int s = f * NSEG + t;
    size_t b0 = ((size_t)s * D_ + d) * 2;
    size_t b1 = (((size_t)SE * D_) + (size_t)s * D_ + d) * 2;
    size_t o  = ((size_t)d * SE + i) * 2;
    cb[o]     = c0 * bank[b0]     + c1 * bank[b1];
    cb[o + 1] = c0 * bank[b0 + 1] + c1 * bank[b1 + 1];
  }
}

// ---------------------------------------------------------------------------
// ecg (B,S,D) -> zero-padded transpose ecgT (B,D,2560) via LDS tile
__global__ void k_pad_transpose(const float* __restrict__ ecg, float* __restrict__ ecgT) {
  __shared__ float tile[32][33];
  int b = blockIdx.z;
  int j0 = blockIdx.x * 32;
  int d0 = blockIdx.y * 32;
  int tx = threadIdx.x, ty = threadIdx.y;
  #pragma unroll
  for (int r = 0; r < 4; r++) {
    int s = j0 + ty + 8 * r - STEPL;
    float v = 0.f;
    if (s >= 0 && s < S_) v = ecg[((size_t)b * S_ + s) * D_ + d0 + tx];
    tile[ty + 8 * r][tx] = v;
  }
  __syncthreads();
  #pragma unroll
  for (int r = 0; r < 4; r++) {
    int d = d0 + ty + 8 * r;
    ecgT[((size_t)b * D_ + d) * PADL + j0 + tx] = tile[tx][ty + 8 * r];
  }
}

// ---------------------------------------------------------------------------
// STFT: one block per (t,d,b) frame. Windowed frame in LDS, 257 bins via
// direct DFT against [n][f]-layout tables (coalesced). Z scaled by 1/256.
__global__ __launch_bounds__(256) void k_stft(const float* __restrict__ ecgT,
                       const float* __restrict__ cNF, const float* __restrict__ sNF,
                       float* __restrict__ ZR, float* __restrict__ ZI) {
  __shared__ float sm[NPER];
  int t = blockIdx.x, d = blockIdx.y, b = blockIdx.z;
  const float* src = ecgT + ((size_t)b * D_ + d) * PADL + t * STEPL;
  for (int n = threadIdx.x; n < NPER; n += 256) {
    float w = 0.5f * (1.0f - cosf(6.2831853071795865f * (float)n / 512.0f));
    sm[n] = src[n] * w;
  }
  __syncthreads();
  size_t base = (((size_t)b * D_ + d) * NSEG + t) * NFREQ;
  for (int f = threadIdx.x; f < NFREQ; f += 256) {
    float ar = 0.f, ai = 0.f;
    for (int n = 0; n < NPER; n++) {
      float x = sm[n];
      ar += x * cNF[n * NFREQ + f];
      ai -= x * sNF[n * NFREQ + f];
    }
    ZR[base + f] = ar * (1.0f / 256.0f);
    ZI[base + f] = ai * (1.0f / 256.0f);
  }
}

// ---------------------------------------------------------------------------
// image rfft (ortho): one block per (f,b), 768 threads over d, block-uniform
// twiddles staged in LDS by the first 196 threads.
__global__ void k_imgfft(const float* __restrict__ img,
                         float* __restrict__ XR, float* __restrict__ XI) {
  __shared__ float cs[NIMG], sn[NIMG];
  int f = blockIdx.x, b = blockIdx.y;
  if (threadIdx.x < NIMG) {
    int m = (threadIdx.x * f) % NIMG;
    double ang = 6.283185307179586476925287 * (double)m / (double)NIMG;
    cs[threadIdx.x] = (float)cos(ang);
    sn[threadIdx.x] = (float)sin(ang);
  }
  __syncthreads();
  int d = threadIdx.x;
  float ar = 0.f, ai = 0.f;
  for (int n = 0; n < NIMG; n++) {
    float x = img[((size_t)b * NIMG + n) * D_ + d];
    ar += x * cs[n];
    ai -= x * sn[n];
  }
  XR[((size_t)b * FIMG + f) * D_ + d] = ar * (1.0f / 14.0f);
  XI[((size_t)b * FIMG + f) * D_ + d] = ai * (1.0f / 14.0f);
}

// ---------------------------------------------------------------------------
// image filter (complex square / 99 * combined bank) + gate-term
__global__ void k_imgfilter(const float* __restrict__ XR, const float* __restrict__ XI,
                            const float* __restrict__ bank, const float* __restrict__ gsel,
                            float* __restrict__ FR, float* __restrict__ FI,
                            float* __restrict__ GT) {
  int i = blockIdx.x * blockDim.x + threadIdx.x;
  if (i >= B_ * FIMG * D_) return;
  int d = i % D_;
  int f = (i / D_) % FIMG;
  float c0 = cosf(PI_F), c1 = cosf(3.0f * PI_F);
  float xr = XR[i], xi = XI[i];
  float pr = (xr * xr - xi * xi) * (1.0f / 99.0f);
  float pi = (2.0f * xr * xi) * (1.0f / 99.0f);
  size_t b0 = ((size_t)f * D_ + d) * 2;
  size_t b1 = (((size_t)FIMG * D_) + (size_t)f * D_ + d) * 2;
  float cbr = c0 * bank[b0] + c1 * bank[b1];
  float cbi = c0 * bank[b0 + 1] + c1 * bank[b1 + 1];
  float fr = pr * cbr - pi * cbi;
  float fi = pr * cbi + pi * cbr;
  FR[i] = fr; FI[i] = fi;
  GT[i] = fr * gsel[b0] - fi * gsel[b0 + 1];   // Re(F * gate_sel)
}

// Serial-f reduction => deterministic FP order (no atomics)
__global__ void k_gatereduce(const float* __restrict__ GT, float* __restrict__ gmean) {
  int b = blockIdx.x, d = threadIdx.x;
  float a = 0.f;
  for (int f = 0; f < FIMG; f++) a += GT[((size_t)b * FIMG + f) * D_ + d];
  gmean[b * D_ + d] = a * (1.0f / 99.0f);
}

// gate = gmean @ gate_w.T + gate_b  (M=8, tiny -> VALU)
__global__ void k_gategemm(const float* __restrict__ gmean, const float* __restrict__ gw,
                           const float* __restrict__ gb, float* __restrict__ gate) {
  int o = blockIdx.x * blockDim.x + threadIdx.x;
  int b = blockIdx.y;
  float a = gb[o];
  const float* m = gmean + b * D_;
  const float* w = gw + (size_t)o * D_;
  for (int d = 0; d < D_; d++) a += m[d] * w[d];
  gate[b * D_ + o] = a;
}

// ---------------------------------------------------------------------------
// Fused filter + gate + ISTFT + overlap-add. One block per (d,b); all 2313
// bins staged in LDS (18.5KB). Thread tid owns window position n0=tid and all
// 8 output samples s = n0 + 256k; the two overlapping frames share one
// twiddle fetch via cos(θ + πf) = (-1)^f cos(θ).
__global__ __launch_bounds__(256) void k_istft(const float* __restrict__ ZR, const float* __restrict__ ZI,
                        const float* __restrict__ cb, const float* __restrict__ gate,
                        const float* __restrict__ cFN, const float* __restrict__ sFN,
                        float* __restrict__ outT) {
  __shared__ float Lr[SE], Li[SE];
  int d = blockIdx.x, b = blockIdx.y;
  float g = gate[b * D_ + d];
  size_t zb = ((size_t)b * D_ + d) * SE;
  const float* cbp = cb + (size_t)d * SE * 2;
  for (int i = threadIdx.x; i < SE; i += 256) {
    float zr = ZR[zb + i], zi = ZI[zb + i];
    float pr = (zr * zr - zi * zi) * (1.0f / 2313.0f);
    float pi = (2.0f * zr * zi) * (1.0f / 2313.0f);
    float cbr = cbp[2 * i], cbi = cbp[2 * i + 1];
    Lr[i] = (pr * cbr - pi * cbi) * g;
    Li[i] = (pr * cbi + pi * cbr) * g;
  }
  __syncthreads();
  int n0 = threadIdx.x;
  float accA[8], accB[8];
  #pragma unroll
  for (int k = 0; k < 8; k++) { accA[k] = 0.f; accB[k] = 0.f; }
  for (int f = 1; f < 256; f++) {
    float c = cFN[f * NPER + n0], s = sFN[f * NPER + n0];
    float sg = (f & 1) ? -1.f : 1.f;
    float cB = c * sg, sB = s * sg;
    #pragma unroll
    for (int k = 0; k < 8; k++) {
      accA[k] += Lr[(k + 1) * NFREQ + f] * c  - Li[(k + 1) * NFREQ + f] * s;
      accB[k] += Lr[k * NFREQ + f]       * cB - Li[k * NFREQ + f]       * sB;
    }
  }
  float wA = 0.5f * (1.0f - cosf(6.2831853071795865f * (float)n0 / 512.0f));
  float wB = 0.5f * (1.0f - cosf(6.2831853071795865f * (float)(n0 + 256) / 512.0f));
  float inorm = 1.0f / (wA * wA + wB * wB);
  float par = (n0 & 1) ? -1.f : 1.f;
  size_t ob = ((size_t)b * D_ + d) * S_;
  #pragma unroll
  for (int k = 0; k < 8; k++) {
    float SA = 0.5f * (Lr[(k + 1) * NFREQ] + par * Lr[(k + 1) * NFREQ + 256] + 2.0f * accA[k]);
    float SB = 0.5f * (Lr[k * NFREQ]       + par * Lr[k * NFREQ + 256]       + 2.0f * accB[k]);
    outT[ob + n0 + 256 * k] = (wA * SA + wB * SB) * inorm;
  }
}

// ---------------------------------------------------------------------------
// image irfft (ortho) + residual -> preact_c (B,196,768)
__global__ void k_imgirfft(const float* __restrict__ FR, const float* __restrict__ FI,
                           const float* __restrict__ img, float* __restrict__ out) {
  __shared__ float cs[FIMG], sn[FIMG];
  int n = blockIdx.x, b = blockIdx.y;
  if (threadIdx.x < FIMG) {
    int m = (threadIdx.x * n) % NIMG;
    double ang = 6.283185307179586476925287 * (double)m / (double)NIMG;
    cs[threadIdx.x] = (float)cos(ang);
    sn[threadIdx.x] = (float)sin(ang);
  }
  __syncthreads();
  int d = threadIdx.x;
  size_t fb = (size_t)b * FIMG * D_ + d;
  float v = FR[fb];                                  // DC (imag ignored)
  float par = (n & 1) ? -1.f : 1.f;
  v += par * FR[fb + (size_t)98 * D_];               // Nyquist (imag ignored)
  for (int f = 1; f < 98; f++)
    v += 2.0f * (FR[fb + (size_t)f * D_] * cs[f] - FI[fb + (size_t)f * D_] * sn[f]);
  size_t oi = ((size_t)b * NIMG + n) * D_ + d;
  out[oi] = img[oi] + v * (1.0f / 14.0f);
}

// ---------------------------------------------------------------------------
// ecgT_t (B,D,S) -> preact_e (B,S,D) with residual add, LDS tile transpose
__global__ void k_transadd(const float* __restrict__ xT, const float* __restrict__ ecg,
                           float* __restrict__ out) {
  __shared__ float tile[32][33];
  int b = blockIdx.z;
  int s0 = blockIdx.x * 32;
  int d0 = blockIdx.y * 32;
  int tx = threadIdx.x, ty = threadIdx.y;
  #pragma unroll
  for (int r = 0; r < 4; r++) {
    int d = d0 + ty + 8 * r;
    tile[ty + 8 * r][tx] = xT[((size_t)b * D_ + d) * S_ + s0 + tx];
  }
  __syncthreads();
  #pragma unroll
  for (int r = 0; r < 4; r++) {
    int s = s0 + ty + 8 * r;
    size_t idx = ((size_t)b * S_ + s) * D_ + d0 + tx;
    out[idx] = tile[tx][ty + 8 * r] + ecg[idx];
  }
}

// ---------------------------------------------------------------------------
// LayerNorm over D=768 (one block per row); optional f16 mirror for WMMA A.
__global__ __launch_bounds__(256) void k_layernorm(const float* __restrict__ X,
                         const float* __restrict__ g, const float* __restrict__ bta,
                         float* __restrict__ outF, f16* __restrict__ outH) {
  __shared__ float r1[256], r2[256];
  size_t row = blockIdx.x;
  const float* x = X + row * D_;
  int tid = threadIdx.x;
  float v0 = x[tid], v1 = x[tid + 256], v2 = x[tid + 512];
  r1[tid] = v0 + v1 + v2;
  r2[tid] = v0 * v0 + v1 * v1 + v2 * v2;
  __syncthreads();
  for (int s = 128; s > 0; s >>= 1) {
    if (tid < s) { r1[tid] += r1[tid + s]; r2[tid] += r2[tid + s]; }
    __syncthreads();
  }
  float mean = r1[0] * (1.0f / 768.0f);
  float var  = r2[0] * (1.0f / 768.0f) - mean * mean;
  float rstd = rsqrtf(var + 1e-5f);
  #pragma unroll
  for (int j = 0; j < 3; j++) {
    int c = tid + 256 * j;
    float v = (x[c] - mean) * rstd * g[c] + bta[c];
    if (outF) outF[row * D_ + c] = v;
    if (outH) outH[row * D_ + c] = (f16)v;
  }
}

// ---------------------------------------------------------------------------
// Async global->LDS copy of one 16-byte chunk (CDNA5, tracked by ASYNCcnt).
__device__ __forceinline__ void async_copy_b128(uint32_t lds_off, const void* gptr) {
  asm volatile("global_load_async_to_lds_b128 %0, %1, off"
               :: "v"(lds_off), "v"(gptr) : "memory");
}
__device__ __forceinline__ void wait_asynccnt0() {
  asm volatile("s_wait_asynccnt 0x0" ::: "memory");
}

// ---------------------------------------------------------------------------
// WMMA GEMM: out[m,n] = act( sum_k A[m,k]*W[n,k] + bias[n] (+resid[m,n]) )
// Block = 8 waves; block tile = 32 rows x 256 cols; each wave owns a 32x32
// output tile (2x2 WMMA accumulators -> 4 WMMAs per 32-wide K step with only
// 2 A + 2 B fragment fetches). The full 32x768 f16 A block is staged once
// into LDS via global_load_async_to_lds_b128 (ASYNCcnt), then the 24-step
// K loop runs barrier-free. LDS rows padded to 776 halfs (1552B == 16B mod
// 256B) so the 16 lanes' ds_load_b128 A-fragment reads cover all 64 banks.
// Fragment layouts per CDNA5 ISA:
//   A (16x32 f16): lane -> M=lane&15, half=lane>>4; e<8 -> K=half*8+e,
//                  e>=8 -> K=16+half*8+(e-8)
//   B (32x16 f16): lane -> N=lane&15, half=lane>>4; e<8 -> K=half*8+e, e>=8 -> K=16+half*8+(e-8)
//   C/D: 8 f32; r -> M = 8*(lane>>4)+r, N = lane&15
__global__ __launch_bounds__(256) void k_gemm_wmma(const f16* __restrict__ A, const f16* __restrict__ W,
                          const float* __restrict__ bias, const float* __restrict__ resid,
                          float* __restrict__ outF, f16* __restrict__ outH, int act) {
  constexpr int RS = 776;                 // padded LDS row stride in halfs
  __shared__ f16 As[32 * RS];             // 48.5 KB
  int wv   = threadIdx.x >> 5;
  int lane = threadIdx.x & 31;
  int m0 = blockIdx.y * 32;
  int n0 = blockIdx.x * 256 + wv * 32;
  int mr = lane & 15, hb = lane >> 4;

  // ---- async-stage the whole 32x768 f16 A block (49152B = 3072 x 16B) ----
  {
    uint32_t lds_base = (uint32_t)(size_t)(&As[0]);
    const char* gbase = (const char*)(A + (size_t)m0 * 768);
    #pragma unroll
    for (int i = 0; i < 12; i++) {
      int c = (int)threadIdx.x + 256 * i;
      int row = c / 96;                     // 96 chunks per 1536B source row
      int off = (c - row * 96) * 16;        // byte offset within row
      uint32_t l = lds_base + (uint32_t)(row * (RS * 2) + off);
      const char* gp = gbase + (size_t)row * 1536 + off;
      async_copy_b128(l, gp);
    }
    wait_asynccnt0();                       // our own copies done
  }
  __syncthreads();                          // everyone's copies visible

  V8F acc[2][2];
  #pragma unroll
  for (int mi = 0; mi < 2; mi++)
    #pragma unroll
    for (int ni = 0; ni < 2; ni++)
      #pragma unroll
      for (int r = 0; r < 8; r++) acc[mi][ni].e[r] = 0.f;

  for (int kk = 0; kk < 768; kk += 32) {
    V16H a[2], bm[2];
    #pragma unroll
    for (int mi = 0; mi < 2; mi++) {
      const f16* ap = &As[(mi * 16 + mr) * RS + kk];
      #pragma unroll
      for (int e = 0; e < 8; e++) a[mi].e[e] = ap[hb * 8 + e];
      #pragma unroll
      for (int e = 0; e < 8; e++) a[mi].e[8 + e] = ap[16 + hb * 8 + e];
    }
    #pragma unroll
    for (int ni = 0; ni < 2; ni++) {
      const f16* wp = W + (size_t)(n0 + ni * 16 + mr) * 768 + kk;
      #pragma unroll
      for (int e = 0; e < 8; e++) bm[ni].e[e] = wp[hb * 8 + e];
      #pragma unroll
      for (int e = 0; e < 8; e++) bm[ni].e[8 + e] = wp[16 + hb * 8 + e];
    }
    #pragma unroll
    for (int mi = 0; mi < 2; mi++)
      #pragma unroll
      for (int ni = 0; ni < 2; ni++)
        acc[mi][ni].v = __builtin_amdgcn_wmma_f32_16x16x32_f16(
            false, a[mi].v, false, bm[ni].v, (short)0, acc[mi][ni].v, false, false);
  }

  #pragma unroll
  for (int mi = 0; mi < 2; mi++)
    #pragma unroll
    for (int ni = 0; ni < 2; ni++) {
      int col = n0 + ni * 16 + mr;
      int rb  = m0 + mi * 16 + hb * 8;
      float bv = bias[col];
      #pragma unroll
      for (int r = 0; r < 8; r++) {
        size_t oi = (size_t)(rb + r) * 768 + col;
        float v = acc[mi][ni].e[r] + bv;
        if (resid) v += resid[oi];
        if (act)   v = 0.5f * v * (1.0f + erff(v * 0.7071067811865475f)); // exact GELU
        if (outF) outF[oi] = v;
        if (outH) outH[oi] = (f16)v;
      }
    }
}

// ---------------------------------------------------------------------------
extern "C" void kernel_launch(void* const* d_in, const int* in_sizes, int n_in,
                              void* d_out, int out_size, void* d_ws, size_t ws_size,
                              hipStream_t stream) {
  const float* ecg    = (const float*)d_in[0];
  const float* image  = (const float*)d_in[1];
  const float* bankE  = (const float*)d_in[2];
  const float* bankC  = (const float*)d_in[3];
  const float* gsel   = (const float*)d_in[4];
  const float* gw     = (const float*)d_in[5];
  const float* gb     = (const float*)d_in[6];
  const float* e_ln1_g = (const float*)d_in[7];
  const float* e_ln1_b = (const float*)d_in[8];
  const float* e_w1    = (const float*)d_in[9];
  const float* e_b1    = (const float*)d_in[10];
  const float* e_w2    = (const float*)d_in[11];
  const float* e_b2    = (const float*)d_in[12];
  const float* e_ln2_g = (const float*)d_in[13];
  const float* e_ln2_b = (const float*)d_in[14];
  const float* c_ln1_g = (const float*)d_in[15];
  const float* c_ln1_b = (const float*)d_in[16];
  const float* c_w1    = (const float*)d_in[17];
  const float* c_b1    = (const float*)d_in[18];
  const float* c_w2    = (const float*)d_in[19];
  const float* c_b2    = (const float*)d_in[20];
  const float* c_ln2_g = (const float*)d_in[21];
  const float* c_ln2_b = (const float*)d_in[22];

  float* out_text = (float*)d_out;                         // (8,2048,768)
  float* out_img  = (float*)d_out + (size_t)B_ * S_ * D_;  // (8,196,768)

  // ---- workspace bump allocation (float granularity, lifetime-aliased) ----
  float* wsf = (float*)d_ws;
  size_t off = 0;
  auto alloc = [&](size_t n) { float* p = wsf + off; off += n; return p; };

  float* slotA  = alloc((size_t)B_ * D_ * PADL);       // ecgT, then ecgT_t
  float* ZR     = alloc((size_t)B_ * D_ * SE);         // then xlnE (f32)
  float* ZI     = alloc((size_t)B_ * D_ * SE);         // then yE
  float* cNF    = alloc(NPER * NFREQ);
  float* sNF    = alloc(NPER * NFREQ);
  float* cFN    = alloc(NPER * NFREQ);
  float* sFN    = alloc(NPER * NFREQ);
  float* cbE    = alloc((size_t)SE * D_ * 2);
  float* XR     = alloc((size_t)B_ * FIMG * D_);
  float* XI     = alloc((size_t)B_ * FIMG * D_);
  float* FRb    = alloc((size_t)B_ * FIMG * D_);
  float* FIb    = alloc((size_t)B_ * FIMG * D_);
  float* GT     = alloc((size_t)B_ * FIMG * D_);
  float* gmean  = alloc(B_ * D_);
  float* gate   = alloc(B_ * D_);
  float* preE   = alloc((size_t)B_ * S_ * D_);         // then hE (f16)
  float* xlnE16f= alloc((size_t)B_ * S_ * D_ / 2);     // f16 storage
  float* preC   = alloc((size_t)B_ * NIMG * D_);
  float* xlnC   = alloc((size_t)B_ * NIMG * D_);
  float* xlnC16f= alloc((size_t)B_ * NIMG * D_ / 2);
  float* hC16f  = alloc((size_t)B_ * NIMG * D_ / 2);
  float* yC     = alloc((size_t)B_ * NIMG * D_);
  float* w1e16f = alloc((size_t)D_ * D_ / 2);
  float* w2e16f = alloc((size_t)D_ * D_ / 2);
  float* w1c16f = alloc((size_t)D_ * D_ / 2);
  float* w2c16f = alloc((size_t)D_ * D_ / 2);

  float* ecgT   = slotA;
  float* ecgTt  = slotA;       // ecgT dead after k_stft
  float* xlnE   = ZR;          // ZR dead after k_istft
  float* yE     = ZI;          // ZI dead after k_istft
  f16*   hE     = (f16*)preE;  // preact_e dead after ln1
  f16*   xlnE16 = (f16*)xlnE16f;
  f16*   xlnC16 = (f16*)xlnC16f;
  f16*   hC     = (f16*)hC16f;
  f16*   w1e16  = (f16*)w1e16f;
  f16*   w2e16  = (f16*)w2e16f;
  f16*   w1c16  = (f16*)w1c16f;
  f16*   w2c16  = (f16*)w2c16f;

  const int WN = D_ * D_;   // 589824

  // ---- weight conversion, tables, combined bank ----
  k_cvt<<<(WN + 255) / 256, 256, 0, stream>>>(e_w1, w1e16, WN);
  k_cvt<<<(WN + 255) / 256, 256, 0, stream>>>(e_w2, w2e16, WN);
  k_cvt<<<(WN + 255) / 256, 256, 0, stream>>>(c_w1, w1c16, WN);
  k_cvt<<<(WN + 255) / 256, 256, 0, stream>>>(c_w2, w2c16, WN);
  k_tables<<<(NPER * NFREQ + 255) / 256, 256, 0, stream>>>(cNF, sNF, cFN, sFN);
  k_cbE<<<D_, 256, 0, stream>>>(bankE, cbE);

  // ---- ecg STFT path ----
  k_pad_transpose<<<dim3(PADL / 32, D_ / 32, B_), dim3(32, 8), 0, stream>>>(ecg, ecgT);
  k_stft<<<dim3(NSEG, D_, B_), 256, 0, stream>>>(ecgT, cNF, sNF, ZR, ZI);

  // ---- image path ----
  k_imgfft<<<dim3(FIMG, B_), D_, 0, stream>>>(image, XR, XI);
  k_imgfilter<<<(B_ * FIMG * D_ + 255) / 256, 256, 0, stream>>>(XR, XI, bankC, gsel, FRb, FIb, GT);
  k_gatereduce<<<B_, D_, 0, stream>>>(GT, gmean);
  k_gategemm<<<dim3(D_ / 256, B_), 256, 0, stream>>>(gmean, gw, gb, gate);

  // ---- fused filter+gate+ISTFT, image irfft+residual ----
  k_istft<<<dim3(D_, B_), 256, 0, stream>>>(ZR, ZI, cbE, gate, cFN, sFN, ecgTt);
  k_imgirfft<<<dim3(NIMG, B_), D_, 0, stream>>>(FRb, FIb, image, preC);
  k_transadd<<<dim3(S_ / 32, D_ / 32, B_), dim3(32, 8), 0, stream>>>(ecgTt, ecg, preE);

  // ---- text addnorm (M = 16384, 32-row x 256-col GEMM blocks) ----
  const int Me = B_ * S_;
  k_layernorm<<<Me, 256, 0, stream>>>(preE, e_ln1_g, e_ln1_b, xlnE, xlnE16);
  k_gemm_wmma<<<dim3(D_ / 256, Me / 32), 256, 0, stream>>>(xlnE16, w1e16, e_b1, nullptr, nullptr, hE, 1);
  k_gemm_wmma<<<dim3(D_ / 256, Me / 32), 256, 0, stream>>>(hE, w2e16, e_b2, xlnE, yE, nullptr, 0);
  k_layernorm<<<Me, 256, 0, stream>>>(yE, e_ln2_g, e_ln2_b, out_text, nullptr);

  // ---- image addnorm (M = 1568 = 32*49) ----
  const int Mc = B_ * NIMG;
  k_layernorm<<<Mc, 256, 0, stream>>>(preC, c_ln1_g, c_ln1_b, xlnC, xlnC16);
  k_gemm_wmma<<<dim3(D_ / 256, Mc / 32), 256, 0, stream>>>(xlnC16, w1c16, c_b1, nullptr, nullptr, hC, 1);
  k_gemm_wmma<<<dim3(D_ / 256, Mc / 32), 256, 0, stream>>>(hC, w2c16, c_b2, xlnC, yC, nullptr, 0);
  k_layernorm<<<Mc, 256, 0, stream>>>(yC, c_ln2_g, c_ln2_b, out_img, nullptr);

  (void)in_sizes; (void)n_in; (void)out_size; (void)ws_size;
}
